// naive_match_model_69200513073212
// MI455X (gfx1250) — compile-verified, matrix operand order checked by reading, not run on hardware
//
#include <hip/hip_runtime.h>

// ---------------------------------------------------------------------------
// ESIM cross-attention + pooling for MI455X (gfx1250, wave32, WMMA).
// B=128, LQ=LA=256, D=384.  All GEMMs via v_wmma_f32_16x16x32_bf16.
// Wave tile 32x64 (2x4 WMMA tiles) -> 8 WMMA per 12 b128 loads per K-step.
// GEMM uses uniform-base + 32-bit lane offsets (GVS addressing) and
// __launch_bounds__(256,1) to avoid accumulator spills.
// ---------------------------------------------------------------------------

typedef __attribute__((ext_vector_type(16))) __bf16        v16bf;
typedef __attribute__((ext_vector_type(8)))  float         v8f;
typedef __attribute__((ext_vector_type(4)))  unsigned int  u32x4;

#define NB   128
#define NLQ  256
#define NLA  256
#define ND   384
#define MAT  (NLQ * NLA)        // 65536 elements per batch score matrix

// ---------------------------- small helpers --------------------------------

__device__ __forceinline__ unsigned short f32_to_bf16_rne(float f) {
  unsigned int u = __float_as_uint(f);
  u += 0x7FFFu + ((u >> 16) & 1u);
  return (unsigned short)(u >> 16);
}

__device__ __forceinline__ float bf16_to_f32(unsigned short h) {
  return __uint_as_float(((unsigned int)h) << 16);
}

// Load one WMMA bf16 operand fragment (16 bf16 per lane = 2 x b128).
// Caller passes the lane-resolved base: row (m0 + lane%16), k0 + 16*(lane/16).
__device__ __forceinline__ v16bf load_frag(const unsigned short* p) {
  union { u32x4 u[2]; v16bf v; } t;
  t.u[0] = *reinterpret_cast<const u32x4*>(p);
  t.u[1] = *reinterpret_cast<const u32x4*>(p + 8);
  return t.v;
}

__device__ __forceinline__ void store_tile_f32(float* C, int ldc, int row0,
                                               int col, const v8f& c) {
#pragma unroll
  for (int v = 0; v < 8; ++v) C[(size_t)(row0 + v) * ldc + col] = c[v];
}

__device__ __forceinline__ void store_tile_bf16(unsigned short* C, int ldc,
                                                int row0, int col, const v8f& c) {
#pragma unroll
  for (int v = 0; v < 8; ++v)
    C[(size_t)(row0 + v) * ldc + col] = f32_to_bf16_rne(c[v]);
}

// ------------------------------ kernels ------------------------------------

__global__ void cvt_f32_bf16_kernel(const float* __restrict__ in,
                                    unsigned short* __restrict__ out, int n) {
  int i = blockIdx.x * blockDim.x + threadIdx.x;
  if (i < n) out[i] = f32_to_bf16_rne(in[i]);
}

// in [B][T][D] bf16 -> out [B][D][T] bf16, 32x32 LDS tiles (+1 pad vs banks)
__global__ void transpose_bf16_kernel(const unsigned short* __restrict__ in,
                                      unsigned short* __restrict__ out,
                                      int T, int D) {
  __shared__ unsigned short tile[32][33];
  const int b = blockIdx.z;
  const unsigned short* ib = in  + (size_t)b * T * D;
  unsigned short*       ob = out + (size_t)b * T * D;
  const int d0 = blockIdx.x * 32;
  const int t0 = blockIdx.y * 32;
  for (int i = threadIdx.y; i < 32; i += 8)
    tile[i][threadIdx.x] = ib[(size_t)(t0 + i) * D + d0 + threadIdx.x];
  __syncthreads();
  for (int i = threadIdx.y; i < 32; i += 8)
    ob[(size_t)(d0 + i) * T + t0 + threadIdx.x] = tile[threadIdx.x][i];
}

// C[M][N] = A[M][K] * Bt[N][K]^T  (bf16 in, f32 accum).
// Wave tile 32(M) x 64(N) = 2x4 WMMA tiles; 8 waves/block arranged 4(M)x2(N)
// => block tile 128x128; grid (N/128, M/128, B).
// Uniform bases + 32-bit per-lane offsets so loads use the saddr (GVS) form.
__global__ void __launch_bounds__(256, 1)
wmma_gemm_bf16_kernel(const unsigned short* __restrict__ A,
                      const unsigned short* __restrict__ Bt,
                      void* __restrict__ Cout,
                      int M, int N, int K, int outBf16) {
  const int b = blockIdx.z;
  const unsigned short* Ab = A  + (size_t)b * M * K;   // uniform base
  const unsigned short* Bb = Bt + (size_t)b * N * K;   // uniform base

  const int lane = threadIdx.x & 31;
  const int wid  = threadIdx.x >> 5;
  const int hl   = lane >> 4;          // half of wave: K sub-chunk select
  const int r    = lane & 15;          // row/col within 16
  const int m0   = blockIdx.y * 128 + (wid & 3) * 32;   // 0,32,64,96
  const int n0   = blockIdx.x * 128 + (wid >> 2) * 64;  // 0,64

  // Per-lane 32-bit element offsets (max < 2^18, fits easily).
  unsigned aOff[2], bOff[4];
#pragma unroll
  for (int i = 0; i < 2; ++i)
    aOff[i] = (unsigned)(m0 + 16 * i + r) * (unsigned)K + 16u * hl;
#pragma unroll
  for (int j = 0; j < 4; ++j)
    bOff[j] = (unsigned)(n0 + 16 * j + r) * (unsigned)K + 16u * hl;

  v8f acc[2][4];
#pragma unroll
  for (int i = 0; i < 2; ++i)
#pragma unroll
    for (int j = 0; j < 4; ++j) acc[i][j] = (v8f){};

  for (int k0 = 0; k0 < K; k0 += 32) {
    v16bf afrag[2], bfrag[4];
#pragma unroll
    for (int i = 0; i < 2; ++i) afrag[i] = load_frag(Ab + aOff[i] + k0);
#pragma unroll
    for (int j = 0; j < 4; ++j) bfrag[j] = load_frag(Bb + bOff[j] + k0);
    if (k0 + 32 < K) {                      // L2 prefetch of next K chunk
      __builtin_prefetch(Ab + aOff[0] + k0 + 32, 0, 1);
      __builtin_prefetch(Bb + bOff[0] + k0 + 32, 0, 1);
    }
#pragma unroll
    for (int i = 0; i < 2; ++i)
#pragma unroll
      for (int j = 0; j < 4; ++j)
        acc[i][j] = __builtin_amdgcn_wmma_f32_16x16x32_bf16(
            false, afrag[i], false, bfrag[j], (short)0, acc[i][j], false, false);
  }

  // C/D layout: VGPR v -> row (v + 8*hl), col = lane%16.
  if (!outBf16) {
    float* C = (float*)Cout + (size_t)b * M * N;
#pragma unroll
    for (int i = 0; i < 2; ++i)
#pragma unroll
      for (int j = 0; j < 4; ++j)
        store_tile_f32(C, N, m0 + 16 * i + 8 * hl, n0 + 16 * j + r, acc[i][j]);
  } else {
    unsigned short* C = (unsigned short*)Cout + (size_t)b * M * N;
#pragma unroll
    for (int i = 0; i < 2; ++i)
#pragma unroll
      for (int j = 0; j < 4; ++j)
        store_tile_bf16(C, N, m0 + 16 * i + 8 * hl, n0 + 16 * j + r, acc[i][j]);
  }
}

// Softmax over 256 elements; one block per (x, b).  rowStride/strideT select
// row-mode (256,1) or column-mode (1,256) reads of S [B][256][256].
// Output always written contiguously: P[b][x][t].
__global__ void softmax256_kernel(const float* __restrict__ S,
                                  unsigned short* __restrict__ P,
                                  int rowStride, int strideT) {
  __shared__ float red[256];
  const int b = blockIdx.y, x = blockIdx.x, t = threadIdx.x;
  const float* src = S + (size_t)b * MAT + (size_t)x * rowStride;
  const float  v   = src[(size_t)t * strideT];

  red[t] = v; __syncthreads();
#pragma unroll
  for (int o = 128; o > 0; o >>= 1) {
    if (t < o) red[t] = fmaxf(red[t], red[t + o]);
    __syncthreads();
  }
  const float mx = red[0]; __syncthreads();

  const float e = __expf(v - mx);
  red[t] = e; __syncthreads();
#pragma unroll
  for (int o = 128; o > 0; o >>= 1) {
    if (t < o) red[t] += red[t + o];
    __syncthreads();
  }
  const float inv = 1.0f / red[0];
  P[(size_t)b * MAT + (size_t)x * 256 + t] = f32_to_bf16_rne(e * inv);
}

// Fused: token-wise l2norm(concat(feat, featI)) -> masked mean -> l2norm.
// One block per batch; thread t owns concat dims {t, t+256, t+512} of 768.
__global__ void pool_norm_kernel(const float* __restrict__ feat,
                                 const unsigned short* __restrict__ featI,
                                 const int* __restrict__ mask,
                                 float* __restrict__ out, int T) {
  __shared__ float red[256];
  const int b = blockIdx.x, t = threadIdx.x;
  const float*          fb = feat  + (size_t)b * T * ND;
  const unsigned short* ib = featI + (size_t)b * T * ND;
  const int*            mb = mask  + (size_t)b * T;

  const int d0 = t, d1 = t + 256, d2 = t + 512;   // d0<384; d2>=512>=384
  float a0 = 0.f, a1 = 0.f, a2 = 0.f, cnt = 0.f;

  for (int q = 0; q < T; ++q) {
    const float v0 = fb[(size_t)q * ND + d0];
    const float v1 = (d1 < ND) ? fb[(size_t)q * ND + d1]
                               : bf16_to_f32(ib[(size_t)q * ND + (d1 - ND)]);
    const float v2 = bf16_to_f32(ib[(size_t)q * ND + (d2 - ND)]);

    red[t] = v0 * v0 + v1 * v1 + v2 * v2; __syncthreads();
#pragma unroll
    for (int o = 128; o > 0; o >>= 1) {
      if (t < o) red[t] += red[t + o];
      __syncthreads();
    }
    const float invn = 1.0f / fmaxf(sqrtf(red[0]), 1e-12f);
    __syncthreads();

    const float m = (float)mb[q];
    a0 += m * v0 * invn; a1 += m * v1 * invn; a2 += m * v2 * invn; cnt += m;
  }

  const float invc = 1.0f / fmaxf(cnt, 1e-9f);
  a0 *= invc; a1 *= invc; a2 *= invc;

  red[t] = a0 * a0 + a1 * a1 + a2 * a2; __syncthreads();
#pragma unroll
  for (int o = 128; o > 0; o >>= 1) {
    if (t < o) red[t] += red[t + o];
    __syncthreads();
  }
  const float inv = 1.0f / fmaxf(sqrtf(red[0]), 1e-12f);

  float* ob = out + (size_t)b * 768;
  ob[d0] = a0 * inv; ob[d1] = a1 * inv; ob[d2] = a2 * inv;
}

// ------------------------------ launcher -----------------------------------

extern "C" void kernel_launch(void* const* d_in, const int* in_sizes, int n_in,
                              void* d_out, int out_size, void* d_ws, size_t ws_size,
                              hipStream_t stream) {
  (void)in_sizes; (void)n_in; (void)out_size; (void)ws_size;
  const float* q_feat = (const float*)d_in[0];
  const float* a_feat = (const float*)d_in[1];
  const int*   q_mask = (const int*)d_in[2];
  const int*   a_mask = (const int*)d_in[3];
  float*       out    = (float*)d_out;

  const size_t nTok  = (size_t)NB * NLQ * ND;       // elements per feature set
  const size_t nMat  = (size_t)NB * MAT;            // elements per score set

  // Workspace layout (all offsets 16B-aligned).
  char* w = (char*)d_ws;
  unsigned short* qb  = (unsigned short*)w;  w += nTok * 2;   // bf16 Q
  unsigned short* ab  = (unsigned short*)w;  w += nTok * 2;   // bf16 A
  unsigned short* qtb = (unsigned short*)w;  w += nTok * 2;   // bf16 Q^T [D][T]
  unsigned short* atb = (unsigned short*)w;  w += nTok * 2;   // bf16 A^T [D][T]
  float*          S   = (float*)w;           w += nMat * 4;   // fp32 scores
  unsigned short* Pq  = (unsigned short*)w;  w += nMat * 2;   // bf16 softmax rows
  unsigned short* Pa  = (unsigned short*)w;  w += nMat * 2;   // bf16 softmax cols
  unsigned short* qi  = (unsigned short*)w;  w += nTok * 2;   // bf16 attended q
  unsigned short* ai  = (unsigned short*)w;  w += nTok * 2;   // bf16 attended a

  // 1) fp32 -> bf16
  const int nT = (int)nTok;
  cvt_f32_bf16_kernel<<<(nT + 255) / 256, 256, 0, stream>>>(q_feat, qb, nT);
  cvt_f32_bf16_kernel<<<(nT + 255) / 256, 256, 0, stream>>>(a_feat, ab, nT);

  // 2) transposed bf16 copies for WMMA B-operands of P@V
  transpose_bf16_kernel<<<dim3(ND / 32, NLQ / 32, NB), dim3(32, 8), 0, stream>>>(
      qb, qtb, NLQ, ND);
  transpose_bf16_kernel<<<dim3(ND / 32, NLA / 32, NB), dim3(32, 8), 0, stream>>>(
      ab, atb, NLA, ND);

  // 3) scores S = Q @ A^T   (M=256, N=256, K=384) — both operands row-major [tok][d]
  wmma_gemm_bf16_kernel<<<dim3(NLA / 128, NLQ / 128, NB), 256, 0, stream>>>(
      qb, ab, S, NLQ, NLA, ND, 0);

  // 4) softmax over a (rows) -> Pq ; softmax over q (cols) -> Pa[b][a][q]
  softmax256_kernel<<<dim3(NLQ, NB), 256, 0, stream>>>(S, Pq, NLA, 1);
  softmax256_kernel<<<dim3(NLA, NB), 256, 0, stream>>>(S, Pa, 1, NLA);

  // 5) attended features: qi = Pq @ A (K=256), ai = Pa @ Q (K=256); bf16 out
  wmma_gemm_bf16_kernel<<<dim3(ND / 128, NLQ / 128, NB), 256, 0, stream>>>(
      Pq, atb, qi, NLQ, ND, NLA, 1);
  wmma_gemm_bf16_kernel<<<dim3(ND / 128, NLA / 128, NB), 256, 0, stream>>>(
      Pa, qtb, ai, NLA, ND, NLQ, 1);

  // 6) fused concat+l2norm+masked-mean+l2norm pooling
  pool_norm_kernel<<<NB, 256, 0, stream>>>(q_feat, qi, q_mask, out, NLQ);
  pool_norm_kernel<<<NB, 256, 0, stream>>>(a_feat, ai, a_mask,
                                           out + (size_t)NB * 768, NLA);
}
